// Decoder_30588757082991
// MI455X (gfx1250) — matrix operand order repaired and checked by reading, not compiled, not run on hardware
//
#include <hip/hip_runtime.h>
#include <math.h>

// Problem dims (match reference)
#define Bb 128
#define Pp 36
#define Ff 2048
#define Aa 1024
#define Ee 1024
#define Dd 1024
#define Vv 10000
#define Ll 30
#define Tt 29

typedef _Float16 v16h __attribute__((ext_vector_type(16)));
typedef _Float16 v8h  __attribute__((ext_vector_type(8)));
typedef float    v8f  __attribute__((ext_vector_type(8)));

// ---------------------------------------------------------------------------
// WMMA fragment loaders (f16, 16x16x32 tile)
// A 16x32: lanes 0-15 -> row M=lane, elems 0..15 = K 0..15
//          lanes 16-31 -> same rows, K 16..31
// B 32x16: lane&15 -> col N, lanes 0-15 hold K 0..15, lanes 16-31 hold K 16..31
// ---------------------------------------------------------------------------
__device__ __forceinline__ v16h ldfragA(const _Float16* __restrict__ p) {
  v8h lo = *(const v8h*)(p);        // 16B
  v8h hi = *(const v8h*)(p + 8);    // 16B
  return __builtin_shufflevector(lo, hi, 0,1,2,3,4,5,6,7,8,9,10,11,12,13,14,15);
}
__device__ __forceinline__ v16h ldfragB(const _Float16* __restrict__ p) {
  v8h lo = *(const v8h*)(p);
  v8h hi = *(const v8h*)(p + 8);
  return __builtin_shufflevector(lo, hi, 0,1,2,3,4,5,6,7,8,9,10,11,12,13,14,15);
}

// Generic fp16 GEMM, f32 accumulate:  C[M,N] = A[M,K] * W[N,K]^T + bias
// Grid: (M/16, ceil(N/128)); block 128 threads = 4 waves; each wave -> 2 N-tiles.
// Optional row mask (dec_len,t): stores 0 for rows with t >= dec_len[row].
__global__ void gemm_f16_wmma(const _Float16* __restrict__ A, int lda,
                              const _Float16* __restrict__ W, int ldw, int Nrows,
                              const float* __restrict__ bias,
                              float* __restrict__ C, long long ldc, int K,
                              const int* __restrict__ dec_len, int t)
{
  const int lane = threadIdx.x & 31;
  const int wave = threadIdx.x >> 5;
  const int m0 = blockIdx.x * 16;
  const int n0 = blockIdx.y * 128 + wave * 32;
  const int l15 = lane & 15;
  const int hiHalf = (lane & 16) ? 1 : 0;

  const int row = m0 + l15;
  const _Float16* Arow = A + (long long)row * lda + (hiHalf ? 16 : 0);

  const int col0 = n0 + l15;
  const int col1 = col0 + 16;
  const int cc0 = col0 < Nrows ? col0 : Nrows - 1;  // clamp for safe OOB loads
  const int cc1 = col1 < Nrows ? col1 : Nrows - 1;
  const _Float16* W0 = W + (long long)cc0 * ldw + (hiHalf ? 16 : 0);
  const _Float16* W1 = W + (long long)cc1 * ldw + (hiHalf ? 16 : 0);

  v8f acc0 = {};
  v8f acc1 = {};
  for (int k0 = 0; k0 < K; k0 += 32) {
    v16h a  = ldfragA(Arow + k0);
    v16h b0 = ldfragB(W0 + k0);
    v16h b1 = ldfragB(W1 + k0);
    acc0 = __builtin_amdgcn_wmma_f32_16x16x32_f16(false, a, false, b0, (short)0, acc0, false, false);
    acc1 = __builtin_amdgcn_wmma_f32_16x16x32_f16(false, a, false, b1, (short)0, acc1, false, false);
  }

  const float bv0 = bias ? bias[cc0] : 0.0f;
  const float bv1 = bias ? bias[cc1] : 0.0f;
  const int mbase = m0 + (hiHalf ? 8 : 0);
#pragma unroll
  for (int r = 0; r < 8; ++r) {
    const int m = mbase + r;
    const bool keep = (dec_len == nullptr) || (t < dec_len[m]);
    if (col0 < Nrows) C[(long long)m * ldc + col0] = keep ? (acc0[r] + bv0) : 0.0f;
    if (col1 < Nrows) C[(long long)m * ldc + col1] = keep ? (acc1[r] + bv1) : 0.0f;
  }
}

// ---------------------------------------------------------------------------
// Prologue kernels
// ---------------------------------------------------------------------------

// Stable descending argsort of caption lengths (B=128, O(B^2) rank), gather
// captions, emit caps / decode_lengths / sort_ind outputs (as floats).
__global__ void sort_kernel(const int* __restrict__ cap_len,
                            const int* __restrict__ caps,
                            int* __restrict__ sort_ind,
                            int* __restrict__ dec_len,
                            int* __restrict__ caps_sorted,
                            float* __restrict__ out_caps,
                            float* __restrict__ out_declen,
                            float* __restrict__ out_sortind)
{
  int i = threadIdx.x;
  int li = cap_len[i];
  int r = 0;
  for (int j = 0; j < Bb; ++j) {
    int lj = cap_len[j];
    if (lj > li || (lj == li && j < i)) ++r;   // stable, descending
  }
  sort_ind[r] = i;
  dec_len[r] = li - 1;
  out_sortind[r] = (float)i;
  out_declen[r] = (float)(li - 1);
  for (int l = 0; l < Ll; ++l) {
    int v = caps[i * Ll + l];
    caps_sorted[r * Ll + l] = v;
    out_caps[r * Ll + l] = (float)v;
  }
}

// Gather feats by sort order into fp16, compute feats_mean directly into
// the xA1 panel (slot [1024,3072)).
__global__ void feats_kernel(const float* __restrict__ image_features,
                             const int* __restrict__ sort_ind,
                             _Float16* __restrict__ feats16,
                             _Float16* __restrict__ xA1)
{
  int b = blockIdx.x, tid = threadIdx.x;
  int src = sort_ind[b];
  for (int f = tid; f < Ff; f += 256) {
    float s = 0.f;
    for (int p = 0; p < Pp; ++p) {
      float v = image_features[((long long)src * Pp + p) * Ff + f];
      feats16[((long long)b * Pp + p) * Ff + f] = (_Float16)v;
      s += v;
    }
    xA1[(long long)b * 5120 + Dd + f] = (_Float16)(s * (1.0f / Pp));
  }
}

// Build fp16 weight panel dst[N][K1+K2] = [s1 | s2] (s2 may be null, K2=0).
__global__ void concat_cvt(const float* __restrict__ s1, int K1,
                           const float* __restrict__ s2, int K2,
                           _Float16* __restrict__ dst, long long total)
{
  long long i = (long long)blockIdx.x * blockDim.x + threadIdx.x;
  if (i >= total) return;
  int K = K1 + K2;
  long long n = i / K;
  int k = (int)(i - n * K);
  float v = (k < K1) ? s1[n * (long long)K1 + k] : s2[n * (long long)K2 + (k - K1)];
  dst[i] = (_Float16)v;
}

__global__ void add_bias(const float* __restrict__ a, const float* __restrict__ b,
                         float* __restrict__ o, int n)
{
  int i = blockIdx.x * blockDim.x + threadIdx.x;
  if (i < n) o[i] = a[i] + b[i];
}

// ---------------------------------------------------------------------------
// Per-step kernels
// ---------------------------------------------------------------------------

// Stage carried h1/h2 and the embedding of token t into the fp16 A-panels.
// xA1 = [h2 | feats_mean | emb_t | h1] (K=5120), xA2 = [awe | h1 | h2] (K=4096)
__global__ void panel_kernel(const float* __restrict__ h1,
                             const float* __restrict__ h2,
                             const float* __restrict__ emb_table,
                             const int* __restrict__ caps_sorted, int t,
                             _Float16* __restrict__ xA1,
                             _Float16* __restrict__ xA2)
{
  int b = blockIdx.x, tid = threadIdx.x;
  int tok = caps_sorted[b * Ll + t];
  const float* e = emb_table + (long long)tok * Ee;
  _Float16* p1 = xA1 + (long long)b * 5120;
  _Float16* p2 = xA2 + (long long)b * 4096;
  for (int i = tid; i < Dd; i += 256) {
    _Float16 h2v = (_Float16)h2[b * Dd + i];
    p1[i] = h2v;                                  // x1: h2
    p1[3072 + i] = (_Float16)e[i];                // x1: emb_t
    p1[4096 + i] = (_Float16)h1[b * Dd + i];      // h1 for W_hh
    p2[3072 + i] = h2v;                           // h2 for lm W_hh
  }
}

__device__ __forceinline__ float sigmoidf(float x) { return 1.0f / (1.0f + expf(-x)); }

// LSTM cell from gate pre-activations. Masked values go to the f32 carries;
// the *unmasked* new h (as the reference uses within-step) goes to fp16 panels.
__global__ void lstm_update(const float* __restrict__ gates,
                            float* __restrict__ h, float* __restrict__ c,
                            _Float16* __restrict__ dst1, int ld1,
                            _Float16* __restrict__ dst2, int ld2,
                            const int* __restrict__ dec_len, int t)
{
  int idx = blockIdx.x * blockDim.x + threadIdx.x;
  if (idx >= Bb * Dd) return;
  int b = idx >> 10, j = idx & (Dd - 1);
  const float* g = gates + (long long)b * (4 * Dd);
  float gi = g[j], gf = g[Dd + j], gg = g[2 * Dd + j], go = g[3 * Dd + j];
  float co = c[idx], ho = h[idx];
  float cn = sigmoidf(gf) * co + sigmoidf(gi) * tanhf(gg);
  float hn = sigmoidf(go) * tanhf(cn);
  bool m = t < dec_len[b];
  c[idx] = m ? cn : co;
  h[idx] = m ? hn : ho;
  _Float16 hv = (_Float16)hn;                     // unmasked for within-step use
  dst1[(long long)b * ld1 + j] = hv;
  if (dst2) dst2[(long long)b * ld2 + j] = hv;
}

// scores = tanh(at1 + at2) . att3_w + att3_b ; softmax over P ; awe = alpha.feats
__global__ void attn_softmax_awe(const float* __restrict__ at1,
                                 const float* __restrict__ at2,
                                 const _Float16* __restrict__ feats16,
                                 const float* __restrict__ att3_w,
                                 const float* __restrict__ att3_b,
                                 _Float16* __restrict__ awe_slot, int ld_awe)
{
  const int b = blockIdx.x;
  const int tid = threadIdx.x;
  __shared__ float red[256];
  __shared__ float sc[Pp];
  __shared__ float alpha[Pp];
  const float* a2 = at2 + (long long)b * Aa;
  for (int p = 0; p < Pp; ++p) {
    const float* a1 = at1 + ((long long)b * Pp + p) * Aa;
    float s = 0.f;
    for (int a = tid; a < Aa; a += 256) s += tanhf(a1[a] + a2[a]) * att3_w[a];
    red[tid] = s; __syncthreads();
    for (int o = 128; o > 0; o >>= 1) { if (tid < o) red[tid] += red[tid + o]; __syncthreads(); }
    if (tid == 0) sc[p] = red[0] + att3_b[0];
    __syncthreads();
  }
  if (tid == 0) {
    float mx = sc[0];
    for (int p = 1; p < Pp; ++p) mx = fmaxf(mx, sc[p]);
    float sum = 0.f;
    for (int p = 0; p < Pp; ++p) { alpha[p] = expf(sc[p] - mx); sum += alpha[p]; }
    float inv = 1.f / sum;
    for (int p = 0; p < Pp; ++p) alpha[p] *= inv;
  }
  __syncthreads();
  _Float16* awe = awe_slot + (long long)b * ld_awe;
  for (int f = tid; f < Ff; f += 256) {
    float s = 0.f;
    for (int p = 0; p < Pp; ++p)
      s += alpha[p] * (float)feats16[((long long)b * Pp + p) * Ff + f];
    awe[f] = (_Float16)s;
  }
}

// ---------------------------------------------------------------------------
// Host driver (graph-capture safe: only kernel launches + hipMemsetAsync)
// ---------------------------------------------------------------------------
extern "C" void kernel_launch(void* const* d_in, const int* in_sizes, int n_in,
                              void* d_out, int out_size, void* d_ws, size_t ws_size,
                              hipStream_t stream)
{
  (void)in_sizes; (void)n_in; (void)out_size; (void)ws_size;

  const float* image_features   = (const float*)d_in[0];
  const int*   encoded_captions = (const int*)d_in[1];
  const int*   caption_lengths  = (const int*)d_in[2];
  const float* embedding        = (const float*)d_in[3];
  const float* td_w_ih = (const float*)d_in[4];
  const float* td_w_hh = (const float*)d_in[5];
  const float* td_b_ih = (const float*)d_in[6];
  const float* td_b_hh = (const float*)d_in[7];
  const float* lm_w_ih = (const float*)d_in[8];
  const float* lm_w_hh = (const float*)d_in[9];
  const float* lm_b_ih = (const float*)d_in[10];
  const float* lm_b_hh = (const float*)d_in[11];
  const float* att1_w  = (const float*)d_in[12];
  const float* att1_b  = (const float*)d_in[13];
  const float* att2_w  = (const float*)d_in[14];
  const float* att2_b  = (const float*)d_in[15];
  const float* att3_w  = (const float*)d_in[16];
  const float* att3_b  = (const float*)d_in[17];
  const float* word_w  = (const float*)d_in[18];
  const float* word_b  = (const float*)d_in[19];

  float* out = (float*)d_out;
  float* out_preds   = out;                                   // (B,T,V)
  float* out_caps    = out + (size_t)Bb * Tt * Vv;            // (B,L)
  float* out_declen  = out_caps + (size_t)Bb * Ll;            // (B,)
  float* out_sortind = out_declen + Bb;                       // (B,)

  // ---- carve workspace (~150 MB) ----
  char* ws = (char*)d_ws;
  size_t off = 0;
  auto alloc = [&](size_t bytes) -> char* {
    char* p = ws + off;
    off = (off + bytes + 255) & ~(size_t)255;
    return p;
  };
  _Float16* B1w     = (_Float16*)alloc((size_t)4096 * 5120 * 2);   // [td_w_ih|td_w_hh]
  _Float16* B2w     = (_Float16*)alloc((size_t)4096 * 4096 * 2);   // [lm_w_ih|lm_w_hh]
  _Float16* Wword   = (_Float16*)alloc((size_t)Vv * 1024 * 2);
  _Float16* Watt1   = (_Float16*)alloc((size_t)1024 * 2048 * 2);
  _Float16* Watt2   = (_Float16*)alloc((size_t)1024 * 1024 * 2);
  _Float16* feats16 = (_Float16*)alloc((size_t)Bb * Pp * Ff * 2);
  float*    at1     = (float*)alloc((size_t)Bb * Pp * Aa * 4);
  _Float16* xA1     = (_Float16*)alloc((size_t)Bb * 5120 * 2);
  _Float16* xA2     = (_Float16*)alloc((size_t)Bb * 4096 * 2);
  _Float16* h2pred  = (_Float16*)alloc((size_t)Bb * Dd * 2);
  float*    gates1  = (float*)alloc((size_t)Bb * 4096 * 4);
  float*    gates2  = (float*)alloc((size_t)Bb * 4096 * 4);
  float*    at2     = (float*)alloc((size_t)Bb * Aa * 4);
  float*    h1      = (float*)alloc((size_t)Bb * Dd * 4);
  float*    c1      = (float*)alloc((size_t)Bb * Dd * 4);
  float*    h2      = (float*)alloc((size_t)Bb * Dd * 4);
  float*    c2      = (float*)alloc((size_t)Bb * Dd * 4);
  float*    b1      = (float*)alloc(4096 * 4);
  float*    b2      = (float*)alloc(4096 * 4);
  int*      sort_ind    = (int*)alloc(Bb * 4);
  int*      dec_len     = (int*)alloc(Bb * 4);
  int*      caps_sorted = (int*)alloc((size_t)Bb * Ll * 4);

  auto launch_gemm = [&](const _Float16* A, int lda, const _Float16* Wm, int ldw,
                         int Nr, const float* bias, float* C, long long ldc,
                         int M, int K, const int* dl, int t) {
    dim3 grid(M / 16, (Nr + 127) / 128);
    gemm_f16_wmma<<<grid, 128, 0, stream>>>(A, lda, Wm, ldw, Nr, bias, C, ldc, K, dl, t);
  };

  // ---- prologue ----
  sort_kernel<<<1, Bb, 0, stream>>>(caption_lengths, encoded_captions, sort_ind,
                                    dec_len, caps_sorted, out_caps, out_declen, out_sortind);
  feats_kernel<<<Bb, 256, 0, stream>>>(image_features, sort_ind, feats16, xA1);
  { long long tot = (long long)4096 * 5120;
    concat_cvt<<<(unsigned)((tot + 255) / 256), 256, 0, stream>>>(td_w_ih, 4096, td_w_hh, 1024, B1w, tot); }
  { long long tot = (long long)4096 * 4096;
    concat_cvt<<<(unsigned)((tot + 255) / 256), 256, 0, stream>>>(lm_w_ih, 3072, lm_w_hh, 1024, B2w, tot); }
  { long long tot = (long long)1024 * 2048;
    concat_cvt<<<(unsigned)((tot + 255) / 256), 256, 0, stream>>>(att1_w, 2048, nullptr, 0, Watt1, tot); }
  { long long tot = (long long)1024 * 1024;
    concat_cvt<<<(unsigned)((tot + 255) / 256), 256, 0, stream>>>(att2_w, 1024, nullptr, 0, Watt2, tot); }
  { long long tot = (long long)Vv * 1024;
    concat_cvt<<<(unsigned)((tot + 255) / 256), 256, 0, stream>>>(word_w, 1024, nullptr, 0, Wword, tot); }
  add_bias<<<16, 256, 0, stream>>>(td_b_ih, td_b_hh, b1, 4096);
  add_bias<<<16, 256, 0, stream>>>(lm_b_ih, lm_b_hh, b2, 4096);
  hipMemsetAsync(h1, 0, (size_t)Bb * Dd * 4, stream);
  hipMemsetAsync(c1, 0, (size_t)Bb * Dd * 4, stream);
  hipMemsetAsync(h2, 0, (size_t)Bb * Dd * 4, stream);
  hipMemsetAsync(c2, 0, (size_t)Bb * Dd * 4, stream);

  // at1 = feats @ att1_w^T + att1_b   (M=4608, N=1024, K=2048)
  launch_gemm(feats16, Ff, Watt1, Ff, Aa, att1_b, at1, Aa, Bb * Pp, Ff, nullptr, 0);

  // ---- decode steps (sequential dependency -> stream-serialized) ----
  for (int t = 0; t < Tt; ++t) {
    panel_kernel<<<Bb, 256, 0, stream>>>(h1, h2, embedding, caps_sorted, t, xA1, xA2);
    // top-down LSTM gates: K fused over [h2|mean|emb|h1]
    launch_gemm(xA1, 5120, B1w, 5120, 4096, b1, gates1, 4096, Bb, 5120, nullptr, 0);
    lstm_update<<<(Bb * Dd) / 256, 256, 0, stream>>>(gates1, h1, c1,
                                                     xA1 + 4096, 5120, xA2 + 2048, 4096,
                                                     dec_len, t);
    // at2 = nh1 @ att2_w^T + att2_b
    launch_gemm(xA1 + 4096, 5120, Watt2, 1024, 1024, att2_b, at2, 1024, Bb, 1024, nullptr, 0);
    attn_softmax_awe<<<Bb, 256, 0, stream>>>(at1, at2, feats16, att3_w, att3_b, xA2, 4096);
    // language LSTM gates: K fused over [awe|nh1|h2]
    launch_gemm(xA2, 4096, B2w, 4096, 4096, b2, gates2, 4096, Bb, 4096, nullptr, 0);
    lstm_update<<<(Bb * Dd) / 256, 256, 0, stream>>>(gates2, h2, c2,
                                                     h2pred, 1024, nullptr, 0,
                                                     dec_len, t);
    // preds = nh2 @ word_w^T + word_b, masked rows -> 0, written into (B,T,V)
    launch_gemm(h2pred, 1024, Wword, 1024, Vv, word_b,
                out_preds + (long long)t * Vv, (long long)Tt * Vv, Bb, 1024, dec_len, t);
  }
}
